// MultiHeadAttention_85925115723936
// MI455X (gfx1250) — matrix operand order repaired and checked by reading, not compiled, not run on hardware
//
#include <hip/hip_runtime.h>
#include <hip/hip_bf16.h>

typedef __attribute__((ext_vector_type(16))) __bf16 v16bf;
typedef __attribute__((ext_vector_type(8)))  __bf16 v8bf;
typedef __attribute__((ext_vector_type(8)))  float  v8f;

#define B_  16
#define T_  1024
#define D_  1024
#define HS_ 64
#define H_  16

// ---------------------------------------------------------------- helpers

__device__ __forceinline__ v8f zero8() {
  v8f z;
#pragma unroll
  for (int j = 0; j < 8; ++j) z[j] = 0.0f;
  return z;
}

__device__ __forceinline__ v8f wmma_bf16(v16bf a, v16bf b, v8f c) {
  // D = A(16x32 bf16) * B(32x16 bf16) + C(16x16 f32)
  return __builtin_amdgcn_wmma_f32_16x16x32_bf16(
      /*neg_a=*/false, a, /*neg_b=*/false, b,
      /*c_mod=*/(short)0, c, /*reuse_a=*/false, /*reuse_b=*/false);
}

// Build A-operand for lane (m = lane&15, kh = lane>>4) from a row-major row
// pointer already offset to (row m, K base + 16*kh): elems 0-15 = K 16kh..16kh+15
// (contiguous k order, matching the B operands).
__device__ __forceinline__ v16bf make_a(const __bf16* p) {
  v8bf lo = *(const v8bf*)p;
  v8bf hi = *(const v8bf*)(p + 8);
  v16bf r;
#pragma unroll
  for (int j = 0; j < 8; ++j) { r[j] = lo[j]; r[j + 8] = hi[j]; }
  return r;
}

__device__ __forceinline__ float rmax16(float v) {
#pragma unroll
  for (int off = 8; off; off >>= 1) v = fmaxf(v, __shfl_xor(v, off, 16));
  return v;
}
__device__ __forceinline__ float rsum16(float v) {
#pragma unroll
  for (int off = 8; off; off >>= 1) v += __shfl_xor(v, off, 16);
  return v;
}

// ---------------------------------------------------------------- stage 0
// Transpose Wq/Wk/Wv -> [HS x D] bf16; fold Wp over H into WpT[e][k] bf16.
__global__ __launch_bounds__(256) void prep_kernel(
    const float* __restrict__ Wq, const float* __restrict__ Wk,
    const float* __restrict__ Wv, const float* __restrict__ Wp,
    __bf16* __restrict__ WtQ, __bf16* __restrict__ WtK,
    __bf16* __restrict__ WtV, __bf16* __restrict__ WpT) {
  int idx = blockIdx.x * blockDim.x + threadIdx.x;   // 4 * 65536 threads
  int seg = idx >> 16;
  int i   = idx & 65535;
  int a   = i >> 6;   // d (or e) : 0..1023
  int h   = i & 63;   // h (or k) : 0..63
  if (seg == 0) {
    WtQ[h * D_ + a] = (__bf16)Wq[a * HS_ + h];
  } else if (seg == 1) {
    WtK[h * D_ + a] = (__bf16)Wk[a * HS_ + h];
  } else if (seg == 2) {
    WtV[h * D_ + a] = (__bf16)Wv[a * HS_ + h];
  } else {
    float s = 0.0f;
#pragma unroll
    for (int hh = 0; hh < H_; ++hh) s += Wp[(size_t)(hh * HS_ + h) * D_ + a];
    WpT[a * HS_ + h] = (__bf16)s;   // WpT[e][k] = sum_h Wp[h*64+k][e]
  }
}

// ---------------------------------------------------------------- stage 1
// Fused Q/K/V projection: one wave computes all three 16x64 output tiles,
// streaming its x rows exactly once (12 WMMAs per 32-wide K chunk).
__global__ __launch_bounds__(256) void qkv_kernel(
    const float* __restrict__ x,
    const __bf16* __restrict__ WtQ, const __bf16* __restrict__ WtK,
    const __bf16* __restrict__ WtV,
    __bf16* __restrict__ Qbf, __bf16* __restrict__ Kbf,
    __bf16* __restrict__ Vt) {
  int wid  = threadIdx.x >> 5;
  int lane = threadIdx.x & 31;
  int rt   = blockIdx.x * 8 + wid;      // 0..1023 row tiles of 16
  int n  = lane & 15;
  int kh = lane >> 4;
  int row0 = rt * 16;

  const __bf16* Wts[3] = {WtQ, WtK, WtV};

  v8f acc[3][4];
#pragma unroll
  for (int w = 0; w < 3; ++w)
#pragma unroll
    for (int t = 0; t < 4; ++t) acc[w][t] = zero8();

  for (int c = 0; c < D_; c += 32) {
    // A: x rows (f32 -> bf16 on the fly), loaded ONCE per chunk
    const float* pa = x + (size_t)(row0 + n) * D_ + c + 16 * kh;
    v16bf a;
#pragma unroll
    for (int j = 0; j < 8; ++j) a[j] = (__bf16)pa[j];
#pragma unroll
    for (int j = 0; j < 8; ++j) a[8 + j] = (__bf16)pa[8 + j];
#pragma unroll
    for (int w = 0; w < 3; ++w) {
#pragma unroll
      for (int t = 0; t < 4; ++t) {
        const __bf16* pb = Wts[w] + (size_t)(t * 16 + n) * D_ + c + 16 * kh;
        v16bf bm = *(const v16bf*)pb;
        acc[w][t] = wmma_bf16(a, bm, acc[w][t]);
      }
    }
  }

  // Q, K row-major bf16
#pragma unroll
  for (int t = 0; t < 4; ++t)
#pragma unroll
    for (int r = 0; r < 8; ++r) {
      Qbf[(size_t)(row0 + r + 8 * kh) * HS_ + t * 16 + n] = (__bf16)acc[0][t][r];
      Kbf[(size_t)(row0 + r + 8 * kh) * HS_ + t * 16 + n] = (__bf16)acc[1][t][r];
    }
  // V transposed: Vt[b][d][s]
  int b = row0 >> 10;
  int s = row0 & 1023;
#pragma unroll
  for (int t = 0; t < 4; ++t)
#pragma unroll
    for (int r = 0; r < 8; ++r)
      Vt[((size_t)b * HS_ + t * 16 + n) * T_ + s + r + 8 * kh] =
          (__bf16)acc[2][t][r];
}

// ---------------------------------------------------------------- stage 2
// Flash attention: one wave per 16-row query tile; 32 keys per iteration.
__global__ __launch_bounds__(256) void attn_kernel(
    const __bf16* __restrict__ Qbf, const __bf16* __restrict__ Kbf,
    const __bf16* __restrict__ Vt, __bf16* __restrict__ HO) {
  __shared__ __bf16 plds[8][16 * 32];   // per-wave private P staging slab

  int wid  = threadIdx.x >> 5;
  int lane = threadIdx.x & 31;
  int rt   = blockIdx.x * 8 + wid;      // 0..1023
  int b    = rt >> 6;
  int qbase = (rt & 63) * 16;
  int n  = lane & 15;
  int kh = lane >> 4;
  __bf16* my = &plds[wid][0];

  // Q tile as two A operands (K = d 0..31 and 32..63)
  const __bf16* qrow = Qbf + ((size_t)b * T_ + qbase + n) * HS_;
  v16bf aq0 = make_a(qrow + 16 * kh);
  v16bf aq1 = make_a(qrow + 32 + 16 * kh);

  v8f acc[4];
#pragma unroll
  for (int t = 0; t < 4; ++t) acc[t] = zero8();
  float m8[8], l8[8];
#pragma unroll
  for (int r = 0; r < 8; ++r) { m8[r] = -1e30f; l8[r] = 0.0f; }

  const float scale = 0.125f;           // 1/sqrt(64)
  int nit = (qbase + 15) / 32 + 1;      // causal: key chunks needed

  for (int it = 0; it < nit; ++it) {
    int kb = it * 32;
    // ---- scores = Q K^T for 32 keys (two 16-key B tiles, K(d)=64 in 2 steps)
    const __bf16* k0 = Kbf + ((size_t)b * T_ + kb + n) * HS_;
    const __bf16* k1 = Kbf + ((size_t)b * T_ + kb + 16 + n) * HS_;
    v16bf bk00 = *(const v16bf*)(k0 + 16 * kh);
    v16bf bk01 = *(const v16bf*)(k0 + 32 + 16 * kh);
    v16bf bk10 = *(const v16bf*)(k1 + 16 * kh);
    v16bf bk11 = *(const v16bf*)(k1 + 32 + 16 * kh);
    v8f c0 = zero8(); c0 = wmma_bf16(aq0, bk00, c0); c0 = wmma_bf16(aq1, bk01, c0);
    v8f c1 = zero8(); c1 = wmma_bf16(aq0, bk10, c1); c1 = wmma_bf16(aq1, bk11, c1);

    // ---- online softmax update (rows align with C-layout registers)
    int j0 = kb + n, j1 = kb + 16 + n;
#pragma unroll
    for (int r = 0; r < 8; ++r) {
      int qr = qbase + r + 8 * kh;
      float s0 = c0[r] * scale; if (j0 > qr) s0 = -1e9f;
      float s1 = c1[r] * scale; if (j1 > qr) s1 = -1e9f;
      float lm = rmax16(fmaxf(s0, s1));
      float nm = fmaxf(m8[r], lm);
      float corr = __expf(m8[r] - nm);
      float p0 = __expf(s0 - nm);
      float p1 = __expf(s1 - nm);
      float rs = rsum16(p0 + p1);
      l8[r] = l8[r] * corr + rs;
      m8[r] = nm;
#pragma unroll
      for (int t = 0; t < 4; ++t) acc[t][r] *= corr;
      // stage P row-major into this wave's LDS slab
      my[(r + 8 * kh) * 32 + n]      = (__bf16)p0;
      my[(r + 8 * kh) * 32 + 16 + n] = (__bf16)p1;
    }
    asm volatile("s_wait_dscnt 0" ::: "memory");   // wave-private LDS RAW

    // ---- P (16x32) as A operand
    v16bf ap = make_a(my + n * 32 + 16 * kh);

    // ---- acc += P * V  (V stored transposed -> contiguous B loads)
#pragma unroll
    for (int t = 0; t < 4; ++t) {
      const __bf16* pv =
          Vt + ((size_t)b * HS_ + t * 16 + n) * T_ + kb + 16 * kh;
      v16bf bv = *(const v16bf*)pv;
      acc[t] = wmma_bf16(ap, bv, acc[t]);
    }
  }

  // ---- normalize and emit head_out (bf16)
#pragma unroll
  for (int t = 0; t < 4; ++t)
#pragma unroll
    for (int r = 0; r < 8; ++r) {
      float o = acc[t][r] / l8[r];
      HO[((size_t)b * T_ + qbase + r + 8 * kh) * HS_ + t * 16 + n] = (__bf16)o;
    }
}

// ---------------------------------------------------------------- stage 3
// out = head_out @ WpT(folded Wp) + bp -> f32.  Each wave covers 4 column
// blocks (256 outputs) per A-tile load to cut head_out re-reads 4x.
__global__ __launch_bounds__(256) void proj_kernel(
    const __bf16* __restrict__ HO, const __bf16* __restrict__ WpT,
    const float* __restrict__ bp, float* __restrict__ out) {
  int wid  = threadIdx.x >> 5;
  int lane = threadIdx.x & 31;
  int rt   = blockIdx.x * 8 + wid;      // 0..1023 row tiles
  int n  = lane & 15;
  int kh = lane >> 4;
  int row0 = rt * 16;

  const __bf16* hrow = HO + (size_t)(row0 + n) * HS_;
  v16bf a0 = make_a(hrow + 16 * kh);
  v16bf a1 = make_a(hrow + 32 + 16 * kh);

#pragma unroll
  for (int cb = 0; cb < 4; ++cb) {
    int eb = blockIdx.y * 256 + cb * 64;
#pragma unroll
    for (int t = 0; t < 4; ++t) {
      int e = eb + t * 16 + n;
      const __bf16* pb = WpT + (size_t)e * HS_;
      v16bf b0 = *(const v16bf*)(pb + 16 * kh);
      v16bf b1 = *(const v16bf*)(pb + 32 + 16 * kh);
      v8f c = zero8();
      c = wmma_bf16(a0, b0, c);
      c = wmma_bf16(a1, b1, c);
      float bias = bp[e];
#pragma unroll
      for (int r = 0; r < 8; ++r)
        out[(size_t)(row0 + r + 8 * kh) * D_ + e] = c[r] + bias;
    }
  }
}

// ---------------------------------------------------------------- launch

extern "C" void kernel_launch(void* const* d_in, const int* in_sizes, int n_in,
                              void* d_out, int out_size, void* d_ws, size_t ws_size,
                              hipStream_t stream) {
  const float* x  = (const float*)d_in[0];
  const float* Wq = (const float*)d_in[1];
  const float* Wk = (const float*)d_in[2];
  const float* Wv = (const float*)d_in[3];
  const float* Wp = (const float*)d_in[4];
  const float* bp = (const float*)d_in[5];
  float* out = (float*)d_out;

  char* ws = (char*)d_ws;
  // workspace layout (bytes)
  const size_t SZ_WT  = (size_t)HS_ * D_ * sizeof(__bf16);       // 128 KB
  const size_t SZ_QK  = (size_t)B_ * T_ * HS_ * sizeof(__bf16);  // 2 MB
  __bf16* WtQ = (__bf16*)(ws + 0);
  __bf16* WtK = (__bf16*)(ws + SZ_WT);
  __bf16* WtV = (__bf16*)(ws + 2 * SZ_WT);
  __bf16* WpT = (__bf16*)(ws + 3 * SZ_WT);
  __bf16* Qbf = (__bf16*)(ws + 4 * SZ_WT);
  __bf16* Kbf = (__bf16*)(ws + 4 * SZ_WT + SZ_QK);
  __bf16* Vt  = (__bf16*)(ws + 4 * SZ_WT + 2 * SZ_QK);
  __bf16* HO  = (__bf16*)(ws + 4 * SZ_WT + 3 * SZ_QK);
  // total = 4*128KB + 4*2MB = ~8.9 MB

  prep_kernel<<<dim3(1024), 256, 0, stream>>>(Wq, Wk, Wv, Wp, WtQ, WtK, WtV, WpT);
  qkv_kernel<<<dim3(128), 256, 0, stream>>>(x, WtQ, WtK, WtV, Qbf, Kbf, Vt);
  attn_kernel<<<dim3(128), 256, 0, stream>>>(Qbf, Kbf, Vt, HO);
  proj_kernel<<<dim3(128, 4), 256, 0, stream>>>(HO, WpT, bp, out);
}
